// GraphEmbedding_76914274337375
// MI455X (gfx1250) — matrix operand order repaired and checked by reading, not compiled, not run on hardware
//
#include <hip/hip_runtime.h>
#include <hip/hip_bf16.h>

// ---------------------------------------------------------------------------
// GCN on geometric graphs, CDNA5 (gfx1250), wave32 + v_wmma_f32_16x16x32_f16.
// An is never materialized: 0/1 mask tiles are regenerated in registers per
// (row-tile, j-chunk) and consumed directly as the WMMA A operand.
// Y = dinv * (H@W) carried as f16 hi/lo pair -> near-fp32 accuracy with two
// f16 WMMAs (f32 accumulate). Aggregation kernel register-blocks 2 row tiles
// per wave (B tile reused by 4 WMMAs) and double-buffers the LDS Y slab with
// GLOBAL_LOAD_ASYNC_TO_LDS_B128 (ASYNCcnt-tracked, no VGPR round trip).
// ---------------------------------------------------------------------------

typedef __attribute__((ext_vector_type(16))) _Float16 v16h;
typedef __attribute__((ext_vector_type(8)))  float    v8f;
typedef __attribute__((ext_vector_type(4)))  int      v4i;

union V16 {
  v16h     h;
  unsigned u[8];
  uint4    q[2];
};

constexpr int NB    = 2048;  // nodes per graph
constexpr int BATCH = 32;    // graphs

#if defined(__AMDGCN__) && \
    __has_builtin(__builtin_amdgcn_global_load_async_to_lds_b128) && \
    __has_builtin(__builtin_amdgcn_s_wait_asynccnt)
#define USE_ASYNC_LDS 1
// Builtin wants v4i pointers: AS1 (prints as __device__) src, AS3 lds dst.
#define GPTR(p) ((__attribute__((address_space(1))) v4i*)(p))
#define LPTR(p) ((__attribute__((address_space(3))) v4i*)(p))
#else
#define USE_ASYNC_LDS 0
#endif

// ---------------------------------------------------------------------------
// Kernel 1: dinv[i] = 1/sqrt(1 + #{j : |p_i - p_j| < 1})
// (indicator includes j==i; +1 models PyG's extra self loop -> diag weight 2)
// grid (NB/128, BATCH), block 128
// ---------------------------------------------------------------------------
__global__ void __launch_bounds__(128)
deg_kernel(const float* __restrict__ pts, float* __restrict__ dinv) {
  const int g = blockIdx.y;
  __shared__ float2 ptsS[NB];
  const float2* pg = (const float2*)pts + (size_t)g * NB;
  for (int x = threadIdx.x; x < NB; x += 128) ptsS[x] = pg[x];
  __syncthreads();

  const int i = blockIdx.x * 128 + threadIdx.x;
  const float2 pi = ptsS[i];
  int cnt = 0;
  for (int j = 0; j < NB; ++j) {            // LDS broadcast reads
    const float2 pj = ptsS[j];
    const float dx = pi.x - pj.x, dy = pi.y - pj.y;
    cnt += (fmaf(dy, dy, dx * dx) < 1.0f) ? 1 : 0;
  }
  dinv[(size_t)g * NB + i] = 1.0f / sqrtf((float)cnt + 1.0f);
}

// ---------------------------------------------------------------------------
// Kernel 2: Y = dinv ⊙ (H @ W), written as f16 hi/lo pair in the WMMA-B
// packed layout: half Y[g][jchunk=NB/32][c=COUT][jmod=32].
// grid (NB/32, BATCH), block 256. tid -> j = tid&31 (coalesced stores),
// channel group cg = tid>>5 walks c in strides of 8 (c uniform per wave).
// ---------------------------------------------------------------------------
template <int CIN, int COUT>
__global__ void __launch_bounds__(256)
transform_kernel(const float* __restrict__ Hin, const float* __restrict__ W,
                 const float* __restrict__ dinv,
                 _Float16* __restrict__ Yhi, _Float16* __restrict__ Ylo) {
  const int g = blockIdx.y;
  const int jt = blockIdx.x;                       // 32-row slab index
  __shared__ float Hs[32 * (CIN + 1)];             // +1 pad: conflict-free rows

  const float* src = Hin + ((size_t)g * NB + (size_t)jt * 32) * CIN;
  for (int x = threadIdx.x; x < 32 * CIN; x += 256) {
    const int j = x / CIN, k = x - j * CIN;
    Hs[j * (CIN + 1) + k] = src[x];
  }
  __syncthreads();

  const int j  = threadIdx.x & 31;
  const int cg = threadIdx.x >> 5;                 // 0..7
  const float dj = dinv[(size_t)g * NB + jt * 32 + j];

  _Float16* yh = Yhi + ((size_t)g * (NB / 32) + jt) * COUT * 32;
  _Float16* yl = Ylo + ((size_t)g * (NB / 32) + jt) * COUT * 32;

  for (int c = cg; c < COUT; c += 8) {
    float acc = 0.0f;
#pragma unroll 8
    for (int k = 0; k < CIN; ++k)
      acc = fmaf(Hs[j * (CIN + 1) + k], W[(size_t)k * COUT + c], acc);
    const float y = acc * dj;
    const _Float16 hi = (_Float16)y;
    const _Float16 lo = (_Float16)(y - (float)hi);
    yh[c * 32 + j] = hi;                           // 64B coalesced per wave
    yl[c * 32 + j] = lo;
  }
}

// ---------------------------------------------------------------------------
// Kernel 3: Z = relu?( dinv_i * (Σ_j mask_ij Y_j + Y_i) + bias )
// grid (NB/64, BATCH), block 128 (4 waves). Wave w: row pair rp = w>>1
// (2 mask A-tiles, 32 rows), channel half ch = w&1 (CT = COUT/32 tiles).
// Each B tile (2x ds_load_b128 per hi/lo) feeds 4 WMMAs. LDS Y slab is
// double-buffered via async global->LDS; per-thread staging offsets are
// hoisted out of the j-chunk loop (no per-load address recomputation).
// ---------------------------------------------------------------------------
template <int COUT, bool RELU>
__global__ void __launch_bounds__(128)
agg_kernel(const float* __restrict__ pts,
           const _Float16* __restrict__ Yhi, const _Float16* __restrict__ Ylo,
           const float* __restrict__ dinv, const float* __restrict__ bias,
           float* __restrict__ out) {
  constexpr int NT   = COUT / 16;                  // channel tiles total
  constexpr int CT   = NT / 2;                     // channel tiles per wave
  constexpr int NJ   = NB / 32;                    // j-chunks
  constexpr int SLAB = COUT * 20;                  // padded dwords per slab
  constexpr int KH   = COUT / 32;                  // b128 per thread per array
  constexpr int KAS  = 2 * KH;                     // async issues per thread/slab
#if USE_ASYNC_LDS
  constexpr int NBUF = 2;
#else
  constexpr int NBUF = 1;
#endif

  const int g    = blockIdx.y;
  const int tid  = threadIdx.x;
  const int lane = tid & 31;
  const int wave = tid >> 5;
  const int m    = lane & 15;                      // A row / B,C,D column
  const int g16  = lane >> 4;                      // K-half selector

  const int chalf   = wave & 1;                    // channel half
  const int rowpair = wave >> 1;                   // which 2 row tiles
  const int rowbase = (blockIdx.x * 4 + rowpair * 2) * 16;

  __shared__ float2   ptsS[NB];                    // 16 KB
  __shared__ unsigned yS[NBUF][2][SLAB];           // [buf][hi|lo][c][20 dw]

  const unsigned* YG[2] = {
      (const unsigned*)Yhi + (size_t)g * NJ * COUT * 16,
      (const unsigned*)Ylo + (size_t)g * NJ * COUT * 16};

  // Per-thread staging geometry, invariant across j-chunks (lives in VGPRs):
  // chunk k covers uint4 index x = tid + k*128; LDS dword off = (x>>2)*20+(x&3)*4.
  int goff[KH], ldsoff[KH];
#pragma unroll
  for (int k = 0; k < KH; ++k) {
    const int x = tid + k * 128;
    goff[k]   = x;
    ldsoff[k] = (x >> 2) * 20 + (x & 3) * 4;
  }

  // Stage slab jc into LDS buffer b (async; VGPR round trip as fallback).
  auto stage = [&](int jc, int b) {
    const uint4* sh = (const uint4*)(YG[0] + (size_t)jc * COUT * 16);
    const uint4* sl = (const uint4*)(YG[1] + (size_t)jc * COUT * 16);
#pragma unroll
    for (int k = 0; k < KH; ++k) {
#if USE_ASYNC_LDS
      __builtin_amdgcn_global_load_async_to_lds_b128(
          GPTR(sh + goff[k]), LPTR(&yS[b][0][ldsoff[k]]), 0, 0);
      __builtin_amdgcn_global_load_async_to_lds_b128(
          GPTR(sl + goff[k]), LPTR(&yS[b][1][ldsoff[k]]), 0, 0);
#else
      *(uint4*)&yS[b][0][ldsoff[k]] = sh[goff[k]];
      *(uint4*)&yS[b][1][ldsoff[k]] = sl[goff[k]];
#endif
    }
  };

  const float2* pg = (const float2*)pts + (size_t)g * NB;
  for (int x = tid; x < NB; x += 128) ptsS[x] = pg[x];

  const float2 pi0 = pg[rowbase + m];              // row points (global reads,
  const float2 pi1 = pg[rowbase + 16 + m];         //  no LDS dependency)

  v8f zero = {};
  v8f acc[2][CT];
#pragma unroll
  for (int r = 0; r < 2; ++r)
#pragma unroll
    for (int t = 0; t < CT; ++t) acc[r][t] = zero;

#if USE_ASYNC_LDS
  stage(0, 0);
#endif

  for (int jc = 0; jc < NJ; ++jc) {
#if USE_ASYNC_LDS
    const int b = jc & 1;
    __syncthreads();                 // all waves done reading buf b^1
    if (jc + 1 < NJ) {
      stage(jc + 1, b ^ 1);          // overlap next slab with this compute
      __builtin_amdgcn_s_wait_asynccnt(KAS);  // slab jc landed (in-order)
    } else {
      __builtin_amdgcn_s_wait_asynccnt(0);
    }
    __syncthreads();                 // slab jc visible to every wave
#else
    const int b = 0;
    __syncthreads();
    stage(jc, 0);
    __syncthreads();
#endif

    // f16 mask A-tiles (16x32 A layout: lane m = row; VGPR v -> K base
    // g16*16+2v, contiguous K half); two row tiles per wave.
    V16 A0, A1;
    const int jb = jc * 32;
#pragma unroll
    for (int v = 0; v < 8; ++v) {
      const int kb = g16 * 16 + 2 * v;
#pragma unroll
      for (int s = 0; s < 2; ++s) {
        const float2 pj = ptsS[jb + kb + s];
        const float dx0 = pi0.x - pj.x, dy0 = pi0.y - pj.y;
        const float dx1 = pi1.x - pj.x, dy1 = pi1.y - pj.y;
        A0.h[2 * v + s] =
            (fmaf(dy0, dy0, dx0 * dx0) < 1.0f) ? (_Float16)1.0f : (_Float16)0.0f;
        A1.h[2 * v + s] =
            (fmaf(dy1, dy1, dx1 * dx1) < 1.0f) ? (_Float16)1.0f : (_Float16)0.0f;
      }
    }

    // B layout: lane m = col, g16 picks K half, VGPR v packs K=2v,2v+1.
#pragma unroll
    for (int t = 0; t < CT; ++t) {
      const int c16 = chalf * CT + t;
      const int cb  = (c16 * 16 + m) * 20 + g16 * 8;   // dword idx, 16B aligned
      V16 Bh, Bl;
      Bh.q[0] = *(const uint4*)&yS[b][0][cb];
      Bh.q[1] = *(const uint4*)&yS[b][0][cb + 4];
      Bl.q[0] = *(const uint4*)&yS[b][1][cb];
      Bl.q[1] = *(const uint4*)&yS[b][1][cb + 4];
      acc[0][t] = __builtin_amdgcn_wmma_f32_16x16x32_f16(
          false, A0.h, false, Bh.h, (short)0, acc[0][t], false, false);
      acc[0][t] = __builtin_amdgcn_wmma_f32_16x16x32_f16(
          false, A0.h, false, Bl.h, (short)0, acc[0][t], false, false);
      acc[1][t] = __builtin_amdgcn_wmma_f32_16x16x32_f16(
          false, A1.h, false, Bh.h, (short)0, acc[1][t], false, false);
      acc[1][t] = __builtin_amdgcn_wmma_f32_16x16x32_f16(
          false, A1.h, false, Bl.h, (short)0, acc[1][t], false, false);
    }
  }

  // Epilogue: + self-loop Y_i, * dinv_i, + bias, relu. C/D layout:
  // row = base + v + 8*g16, col = lane&15 within tile.
  const _Float16* YhiE = Yhi + (size_t)g * NJ * COUT * 32;
  const _Float16* YloE = Ylo + (size_t)g * NJ * COUT * 32;
#pragma unroll
  for (int r = 0; r < 2; ++r) {
#pragma unroll
    for (int t = 0; t < CT; ++t) {
      const int c = (chalf * CT + t) * 16 + m;
      const float bc = bias[c];
#pragma unroll
      for (int v = 0; v < 8; ++v) {
        const int i = rowbase + r * 16 + v + 8 * g16;
        const size_t yidx = (((size_t)(i >> 5)) * COUT + c) * 32 + (i & 31);
        const float yi = (float)YhiE[yidx] + (float)YloE[yidx];
        float z = dinv[(size_t)g * NB + i] * (acc[r][t][v] + yi) + bc;
        if (RELU) z = fmaxf(z, 0.0f);
        out[((size_t)g * NB + i) * COUT + c] = z;
      }
    }
  }
}

// ---------------------------------------------------------------------------
// Host-side orchestration (graph-capture safe: launches only).
// Inputs: points, W1, b1, W2, b2, W3, b3 (all fp32). Output fp32 [B,N,256].
// ---------------------------------------------------------------------------
extern "C" void kernel_launch(void* const* d_in, const int* in_sizes, int n_in,
                              void* d_out, int out_size, void* d_ws, size_t ws_size,
                              hipStream_t stream) {
  (void)in_sizes; (void)n_in; (void)out_size; (void)ws_size;
  const float* pts = (const float*)d_in[0];
  const float* W1  = (const float*)d_in[1];
  const float* b1  = (const float*)d_in[2];
  const float* W2  = (const float*)d_in[3];
  const float* b2  = (const float*)d_in[4];
  const float* W3  = (const float*)d_in[5];
  const float* b3  = (const float*)d_in[6];
  float* out = (float*)d_out;

  char* ws = (char*)d_ws;
  size_t off = 0;
  float* dinv = (float*)(ws + off);
  off += (size_t)BATCH * NB * sizeof(float);
  off = (off + 255) & ~(size_t)255;
  _Float16* Yhi = (_Float16*)(ws + off);
  off += (size_t)BATCH * NB * 256 * sizeof(_Float16);   // 32 MB (max COUT)
  _Float16* Ylo = (_Float16*)(ws + off);
  off += (size_t)BATCH * NB * 256 * sizeof(_Float16);   // 32 MB
  float* H1 = (float*)(ws + off);
  off += (size_t)BATCH * NB * 64 * sizeof(float);       // 16 MB
  float* H2 = (float*)(ws + off);
  off += (size_t)BATCH * NB * 128 * sizeof(float);      // 32 MB

  const dim3 bDeg(128), gDeg(NB / 128, BATCH);
  const dim3 bT(256),  gT(NB / 32, BATCH);
  const dim3 bA(128),  gA(NB / 64, BATCH);

  deg_kernel<<<gDeg, bDeg, 0, stream>>>(pts, dinv);

  // Layer 1: 2 -> 64, relu
  transform_kernel<2, 64><<<gT, bT, 0, stream>>>(pts, W1, dinv, Yhi, Ylo);
  agg_kernel<64, true><<<gA, bA, 0, stream>>>(pts, Yhi, Ylo, dinv, b1, H1);

  // Layer 2: 64 -> 128, relu
  transform_kernel<64, 128><<<gT, bT, 0, stream>>>(H1, W2, dinv, Yhi, Ylo);
  agg_kernel<128, true><<<gA, bA, 0, stream>>>(pts, Yhi, Ylo, dinv, b2, H2);

  // Layer 3: 128 -> 256, no relu, straight to d_out
  transform_kernel<128, 256><<<gT, bT, 0, stream>>>(H2, W3, dinv, Yhi, Ylo);
  agg_kernel<256, false><<<gA, bA, 0, stream>>>(pts, Yhi, Ylo, dinv, b3, out);
}
